// TokenformerMLPAdapter_20779051778285
// MI455X (gfx1250) — matrix operand order repaired and checked by reading, not compiled, not run on hardware
//
#include <hip/hip_runtime.h>

#define T_DIM 2048
#define H_DIM 2048
#define NH 4
#define HD 512
#define RANK 32

typedef __attribute__((ext_vector_type(16))) __bf16 v16bf;
typedef __attribute__((ext_vector_type(8)))  float  v8f;

__device__ __forceinline__ v8f wmma_bf16(v16bf a, v16bf b, v8f c) {
  // (neg_a, A, neg_b, B, c_mod, C, reuse_a, reuse_b)
  return __builtin_amdgcn_wmma_f32_16x16x32_bf16(false, a, false, b, (short)0, c,
                                                 false, false);
}

union BfBits { unsigned short s; __bf16 b; };

// Split f32 into bf16 hi + bf16 lo (truncation split, ~2^-16 rel error kept)
__device__ __forceinline__ void split1(float x, __bf16& hi, __bf16& lo) {
  unsigned int u = __float_as_uint(x);
  BfBits h; h.s = (unsigned short)(u >> 16);
  float hf = __uint_as_float((unsigned int)h.s << 16);
  float l = x - hf;
  BfBits lb; lb.s = (unsigned short)(__float_as_uint(l) >> 16);
  hi = h.b;
  lo = lb.b;
}

__device__ __forceinline__ v8f vzero() {
  v8f z;
#pragma unroll
  for (int i = 0; i < 8; ++i) z[i] = 0.0f;
  return z;
}

union V16U { v16bf v; uint4 u[2]; };

// A fragment 16x32 from a pre-converted bf16 plane (row-major, contiguous K).
// Per lane: elements 0..7 <- K[kh*8..+8), elements 8..15 <- K[16+kh*8..+8)
__device__ __forceinline__ v16bf load_frag_a(const __bf16* __restrict__ row, int kh) {
  V16U x;
  x.u[0] = *reinterpret_cast<const uint4*>(row + kh * 8);
  x.u[1] = *reinterpret_cast<const uint4*>(row + 16 + kh * 8);
  return x.v;
}

// B fragment 32x16 from a bf16 plane contiguous along K:
// per lane elements 0..7 <- K[kh*8..+8), elements 8..15 <- K[16+kh*8..+8)
__device__ __forceinline__ v16bf load_frag_b16(const __bf16* __restrict__ row, int kh) {
  V16U x;
  x.u[0] = *reinterpret_cast<const uint4*>(row + kh * 8);
  x.u[1] = *reinterpret_cast<const uint4*>(row + 16 + kh * 8);
  return x.v;
}

// ---------------------------------------------------------------------------
// Conversion kernels: one-shot f32 -> bf16 hi/lo planes
// ---------------------------------------------------------------------------
__global__ __launch_bounds__(256) void convert_split_kernel(const float* __restrict__ src,
                                                            __bf16* __restrict__ hi,
                                                            __bf16* __restrict__ lo,
                                                            int nelem) {
  int i = (blockIdx.x * blockDim.x + threadIdx.x) * 4;
  if (i >= nelem) return;
  float4 x = *reinterpret_cast<const float4*>(src + i);
  union { __bf16 b[4]; uint2 u; } H, L;
  __bf16 h, l;
  split1(x.x, h, l); H.b[0] = h; L.b[0] = l;
  split1(x.y, h, l); H.b[1] = h; L.b[1] = l;
  split1(x.z, h, l); H.b[2] = h; L.b[2] = l;
  split1(x.w, h, l); H.b[3] = h; L.b[3] = l;
  *reinterpret_cast<uint2*>(hi + i) = H.u;
  *reinterpret_cast<uint2*>(lo + i) = L.u;
}

// tok_v[k, h*16384 + d*32 + rr]  ->  vT[((k*NH+h)*RANK + rr)*HD + d]  (K=d contiguous)
__global__ __launch_bounds__(256) void convert_v_kernel(const float* __restrict__ tok_v,
                                                        __bf16* __restrict__ vhi,
                                                        __bf16* __restrict__ vlo) {
  int idx = blockIdx.x * blockDim.x + threadIdx.x;
  if (idx >= NH * NH * RANK * HD) return;
  int d = idx & (HD - 1);
  int rest = idx >> 9;          // (k*NH+h)*RANK + rr
  int rr = rest & (RANK - 1);
  int hk = rest >> 5;           // k*NH + h
  int k = hk >> 2;
  int h = hk & 3;
  float x = tok_v[(size_t)k * (H_DIM * RANK) + (size_t)h * (HD * RANK) +
                  (size_t)d * RANK + rr];
  __bf16 hb, lb;
  split1(x, hb, lb);
  vhi[idx] = hb;
  vlo[idx] = lb;
}

// tok_p[rr, j] -> pT[j*RANK + rr]  (K=rr contiguous)
__global__ __launch_bounds__(256) void convert_p_kernel(const float* __restrict__ tok_p,
                                                        __bf16* __restrict__ phi,
                                                        __bf16* __restrict__ plo) {
  int idx = blockIdx.x * blockDim.x + threadIdx.x;
  if (idx >= H_DIM * RANK) return;
  int j = idx >> 5;
  int rr = idx & (RANK - 1);
  float x = tok_p[(size_t)rr * H_DIM + j];
  __bf16 hb, lb;
  split1(x, hb, lb);
  phi[idx] = hb;
  plo[idx] = lb;
}

// ---------------------------------------------------------------------------
// Kernel 1: tiny SDPA scores + softmax over 4 keys. One thread per (t, h).
// ---------------------------------------------------------------------------
__global__ __launch_bounds__(256) void attn_kernel(const float* __restrict__ q,
                                                   const float* __restrict__ tok_k,
                                                   float* __restrict__ attn) {
  int idx = blockIdx.x * blockDim.x + threadIdx.x;
  if (idx >= T_DIM * NH) return;
  int t = idx >> 2;
  int h = idx & 3;
  const float* qr = q + (size_t)t * H_DIM + h * HD;
  const float* k0 = tok_k + 0 * H_DIM + h * HD;
  const float* k1 = tok_k + 1 * H_DIM + h * HD;
  const float* k2 = tok_k + 2 * H_DIM + h * HD;
  const float* k3 = tok_k + 3 * H_DIM + h * HD;
  float s0 = 0.f, s1 = 0.f, s2 = 0.f, s3 = 0.f;
  for (int d = 0; d < HD; d += 4) {
    float4 qv = *reinterpret_cast<const float4*>(qr + d);
    float4 a = *reinterpret_cast<const float4*>(k0 + d);
    float4 b = *reinterpret_cast<const float4*>(k1 + d);
    float4 c = *reinterpret_cast<const float4*>(k2 + d);
    float4 e = *reinterpret_cast<const float4*>(k3 + d);
    s0 += qv.x * a.x + qv.y * a.y + qv.z * a.z + qv.w * a.w;
    s1 += qv.x * b.x + qv.y * b.y + qv.z * b.z + qv.w * b.w;
    s2 += qv.x * c.x + qv.y * c.y + qv.z * c.z + qv.w * c.w;
    s3 += qv.x * e.x + qv.y * e.y + qv.z * e.z + qv.w * e.w;
  }
  const float scale = 0.04419417382415922f;  // 1/sqrt(512)
  s0 *= scale; s1 *= scale; s2 *= scale; s3 *= scale;
  float mx = fmaxf(fmaxf(s0, s1), fmaxf(s2, s3));
  float e0 = __expf(s0 - mx), e1 = __expf(s1 - mx);
  float e2 = __expf(s2 - mx), e3 = __expf(s3 - mx);
  float inv = 1.0f / (e0 + e1 + e2 + e3);
  float* o = attn + (size_t)t * 16 + h * 4;
  o[0] = e0 * inv; o[1] = e1 * inv; o[2] = e2 * inv; o[3] = e3 * inv;
}

// ---------------------------------------------------------------------------
// Kernel 2: coef[t,rr] = sum_{h,k} attn[t,h,k] * (q_slice(T x 512) @ V_hk(512 x 32))
// One block per 16-token tile; 4 waves, wave w handles head h=w; LDS reduce.
// Emits coef as bf16 hi/lo planes for the fused epilogue in kernel 3.
// ---------------------------------------------------------------------------
__global__ __launch_bounds__(128) void coef_kernel(const __bf16* __restrict__ qhi,
                                                   const __bf16* __restrict__ qlo,
                                                   const __bf16* __restrict__ vthi,
                                                   const __bf16* __restrict__ vtlo,
                                                   const float* __restrict__ attn,
                                                   __bf16* __restrict__ coef_hi,
                                                   __bf16* __restrict__ coef_lo) {
  __shared__ float red[NH * 16 * RANK];  // 2048 floats = 8KB
  const int t0 = blockIdx.x * 16;
  const int h = threadIdx.x >> 5;
  const int lane = threadIdx.x & 31;
  const int n = lane & 15;
  const int kh = lane >> 4;

  v8f c0 = vzero(), c1 = vzero();
  const __bf16* arh = qhi + (size_t)(t0 + n) * H_DIM + h * HD;
  const __bf16* arl = qlo + (size_t)(t0 + n) * H_DIM + h * HD;

  for (int k = 0; k < NH; ++k) {
    float wrow[8];
#pragma unroll
    for (int v = 0; v < 8; ++v)
      wrow[v] = attn[(size_t)(t0 + v + kh * 8) * 16 + h * 4 + k];

    const size_t hk = (size_t)(k * NH + h) * RANK;
    const __bf16* vbh = vthi + hk * HD;
    const __bf16* vbl = vtlo + hk * HD;
    v8f u0 = vzero(), u1 = vzero();
    for (int ks = 0; ks < HD / 32; ++ks) {
      v16bf ah = load_frag_a(arh + ks * 32, kh);
      v16bf al = load_frag_a(arl + ks * 32, kh);
      v16bf b0h = load_frag_b16(vbh + (size_t)n * HD + ks * 32, kh);
      v16bf b0l = load_frag_b16(vbl + (size_t)n * HD + ks * 32, kh);
      v16bf b1h = load_frag_b16(vbh + (size_t)(16 + n) * HD + ks * 32, kh);
      v16bf b1l = load_frag_b16(vbl + (size_t)(16 + n) * HD + ks * 32, kh);
      u0 = wmma_bf16(ah, b0h, u0);
      u0 = wmma_bf16(ah, b0l, u0);
      u0 = wmma_bf16(al, b0h, u0);
      u1 = wmma_bf16(ah, b1h, u1);
      u1 = wmma_bf16(ah, b1l, u1);
      u1 = wmma_bf16(al, b1h, u1);
    }
#pragma unroll
    for (int v = 0; v < 8; ++v) {
      c0[v] += wrow[v] * u0[v];
      c1[v] += wrow[v] * u1[v];
    }
  }

#pragma unroll
  for (int v = 0; v < 8; ++v) {
    int m = v + kh * 8;
    red[h * (16 * RANK) + m * RANK + n] = c0[v];
    red[h * (16 * RANK) + m * RANK + 16 + n] = c1[v];
  }
  __syncthreads();
  for (int i = threadIdx.x; i < 16 * RANK; i += 128) {
    float s = red[i] + red[512 + i] + red[1024 + i] + red[1536 + i];
    __bf16 hb, lb;
    split1(s, hb, lb);
    coef_hi[(size_t)t0 * RANK + i] = hb;
    coef_lo[(size_t)t0 * RANK + i] = lb;
  }
}

// ---------------------------------------------------------------------------
// Kernel 3: out = query @ w_base^T + b_base + coef @ tok_p
// Block tile 64(M) x 128(N); 8 waves, each 32x32 (2x2 WMMA fragments).
// Pure b128 loads + WMMA in the K-loop (operands pre-split to bf16 planes).
// ---------------------------------------------------------------------------
__global__ __launch_bounds__(256) void base_kernel(const __bf16* __restrict__ qhi,
                                                   const __bf16* __restrict__ qlo,
                                                   const __bf16* __restrict__ whi,
                                                   const __bf16* __restrict__ wlo,
                                                   const float* __restrict__ bias,
                                                   const __bf16* __restrict__ pthi,
                                                   const __bf16* __restrict__ ptlo,
                                                   const __bf16* __restrict__ coef_hi,
                                                   const __bf16* __restrict__ coef_lo,
                                                   float* __restrict__ out) {
  const int wave = threadIdx.x >> 5;
  const int lane = threadIdx.x & 31;
  const int n = lane & 15;
  const int kh = lane >> 4;
  const int mw = wave >> 2;  // 0..1
  const int nw = wave & 3;   // 0..3
  const int m0 = blockIdx.y * 64 + mw * 32;
  const int n0 = blockIdx.x * 128 + nw * 32;

  v8f acc[2][2];
#pragma unroll
  for (int a = 0; a < 2; ++a)
#pragma unroll
    for (int b = 0; b < 2; ++b) acc[a][b] = vzero();

  const __bf16* ah0 = qhi + (size_t)(m0 + n) * H_DIM;
  const __bf16* ah1 = qhi + (size_t)(m0 + 16 + n) * H_DIM;
  const __bf16* al0 = qlo + (size_t)(m0 + n) * H_DIM;
  const __bf16* al1 = qlo + (size_t)(m0 + 16 + n) * H_DIM;
  const __bf16* bh0 = whi + (size_t)(n0 + n) * H_DIM;
  const __bf16* bh1 = whi + (size_t)(n0 + 16 + n) * H_DIM;
  const __bf16* bl0 = wlo + (size_t)(n0 + n) * H_DIM;
  const __bf16* bl1 = wlo + (size_t)(n0 + 16 + n) * H_DIM;

  for (int kb = 0; kb < H_DIM; kb += 32) {
    if (kb + 32 < H_DIM) {  // global_prefetch_b8 hints for the next K-slice
      __builtin_prefetch(ah0 + kb + 32, 0, 0);
      __builtin_prefetch(bh0 + kb + 32, 0, 0);
    }
    v16bf ah[2], al[2], bh[2], bl[2];
    ah[0] = load_frag_a(ah0 + kb, kh);
    ah[1] = load_frag_a(ah1 + kb, kh);
    al[0] = load_frag_a(al0 + kb, kh);
    al[1] = load_frag_a(al1 + kb, kh);
    bh[0] = load_frag_b16(bh0 + kb, kh);
    bh[1] = load_frag_b16(bh1 + kb, kh);
    bl[0] = load_frag_b16(bl0 + kb, kh);
    bl[1] = load_frag_b16(bl1 + kb, kh);
#pragma unroll
    for (int mt = 0; mt < 2; ++mt)
#pragma unroll
      for (int nt = 0; nt < 2; ++nt) {
        acc[mt][nt] = wmma_bf16(ah[mt], bh[nt], acc[mt][nt]);
        acc[mt][nt] = wmma_bf16(ah[mt], bl[nt], acc[mt][nt]);
        acc[mt][nt] = wmma_bf16(al[mt], bh[nt], acc[mt][nt]);
      }
  }

  // Fused low-rank adapter epilogue: acc += coef(16x32) @ tok_p(32x16), K=32
  {
    v16bf ch[2], cl[2], ph[2], pl[2];
    ch[0] = load_frag_a(coef_hi + (size_t)(m0 + n) * RANK, kh);
    ch[1] = load_frag_a(coef_hi + (size_t)(m0 + 16 + n) * RANK, kh);
    cl[0] = load_frag_a(coef_lo + (size_t)(m0 + n) * RANK, kh);
    cl[1] = load_frag_a(coef_lo + (size_t)(m0 + 16 + n) * RANK, kh);
    ph[0] = load_frag_b16(pthi + (size_t)(n0 + n) * RANK, kh);
    ph[1] = load_frag_b16(pthi + (size_t)(n0 + 16 + n) * RANK, kh);
    pl[0] = load_frag_b16(ptlo + (size_t)(n0 + n) * RANK, kh);
    pl[1] = load_frag_b16(ptlo + (size_t)(n0 + 16 + n) * RANK, kh);
#pragma unroll
    for (int mt = 0; mt < 2; ++mt)
#pragma unroll
      for (int nt = 0; nt < 2; ++nt) {
        acc[mt][nt] = wmma_bf16(ch[mt], ph[nt], acc[mt][nt]);
        acc[mt][nt] = wmma_bf16(ch[mt], pl[nt], acc[mt][nt]);
        acc[mt][nt] = wmma_bf16(cl[mt], ph[nt], acc[mt][nt]);
      }
  }

  // bias + store (C layout: lane holds column n, rows v + kh*8)
#pragma unroll
  for (int nt = 0; nt < 2; ++nt) {
    const int j = n0 + nt * 16 + n;
    const float bb = bias[j];
#pragma unroll
    for (int mt = 0; mt < 2; ++mt) {
#pragma unroll
      for (int v = 0; v < 8; ++v) {
        const int t = m0 + mt * 16 + kh * 8 + v;
        out[(size_t)t * H_DIM + j] = acc[mt][nt][v] + bb;
      }
    }
  }
}

// ---------------------------------------------------------------------------
// Workspace layout (bytes, all offsets 256B aligned). Total ~33.7 MB.
// ---------------------------------------------------------------------------
static const size_t ATTN_OFF  = 0;                        // 131072 (f32 T*16)
static const size_t COEFH_OFF = 131072;                   // 131072 (bf16 T*32)
static const size_t COEFL_OFF = 262144;                   // 131072
static const size_t QHI_OFF   = 393216;                   // 8388608 (bf16 T*H)
static const size_t QLO_OFF   = 8781824;                  // 8388608
static const size_t WHI_OFF   = 17170432;                 // 8388608
static const size_t WLO_OFF   = 25559040;                 // 8388608
static const size_t VTHI_OFF  = 33947648;                 // 524288 (bf16 16*32*512)
static const size_t VTLO_OFF  = 34471936;                 // 524288
static const size_t PTHI_OFF  = 34996224;                 // 131072 (bf16 H*32)
static const size_t PTLO_OFF  = 35127296;                 // 131072 -> total 35258368

extern "C" void kernel_launch(void* const* d_in, const int* in_sizes, int n_in,
                              void* d_out, int out_size, void* d_ws, size_t ws_size,
                              hipStream_t stream) {
  (void)in_sizes; (void)n_in; (void)out_size; (void)ws_size;
  const float* q      = (const float*)d_in[0];
  const float* w_base = (const float*)d_in[1];
  const float* b_base = (const float*)d_in[2];
  const float* tok_k  = (const float*)d_in[3];
  const float* tok_v  = (const float*)d_in[4];
  const float* tok_p  = (const float*)d_in[5];
  float* out = (float*)d_out;

  char* ws = (char*)d_ws;
  float*  attn    = (float*)(ws + ATTN_OFF);
  __bf16* coef_hi = (__bf16*)(ws + COEFH_OFF);
  __bf16* coef_lo = (__bf16*)(ws + COEFL_OFF);
  __bf16* qhi     = (__bf16*)(ws + QHI_OFF);
  __bf16* qlo     = (__bf16*)(ws + QLO_OFF);
  __bf16* whi     = (__bf16*)(ws + WHI_OFF);
  __bf16* wlo     = (__bf16*)(ws + WLO_OFF);
  __bf16* vthi    = (__bf16*)(ws + VTHI_OFF);
  __bf16* vtlo    = (__bf16*)(ws + VTLO_OFF);
  __bf16* pthi    = (__bf16*)(ws + PTHI_OFF);
  __bf16* ptlo    = (__bf16*)(ws + PTLO_OFF);

  const int NQW = T_DIM * H_DIM;  // 4194304
  convert_split_kernel<<<dim3(NQW / 4 / 256), dim3(256), 0, stream>>>(q, qhi, qlo, NQW);
  convert_split_kernel<<<dim3(NQW / 4 / 256), dim3(256), 0, stream>>>(w_base, whi, wlo, NQW);
  convert_v_kernel<<<dim3(NH * NH * RANK * HD / 256), dim3(256), 0, stream>>>(tok_v, vthi, vtlo);
  convert_p_kernel<<<dim3(H_DIM * RANK / 256), dim3(256), 0, stream>>>(tok_p, pthi, ptlo);
  attn_kernel<<<dim3((T_DIM * NH + 255) / 256), dim3(256), 0, stream>>>(q, tok_k, attn);
  coef_kernel<<<dim3(T_DIM / 16), dim3(128), 0, stream>>>(qhi, qlo, vthi, vtlo, attn,
                                                          coef_hi, coef_lo);
  base_kernel<<<dim3(H_DIM / 128, T_DIM / 64), dim3(256), 0, stream>>>(
      qhi, qlo, whi, wlo, b_base, pthi, ptlo, coef_hi, coef_lo, out);
}
